// OnlineTripletSiameseModel_88399016886878
// MI455X (gfx1250) — matrix operand order repaired and checked by reading, not compiled, not run on hardware
//
#include <hip/hip_runtime.h>
#include <hip/hip_bf16.h>

// ---------------------------------------------------------------------------
// Triplet loss: emb = X @ W + b for anchor/pos/neg, squared distances, hinge,
// mean.  B=8192, K=2304 (48*48), EMB=128.
//
// MI455X plan: HBM-bound (226 MB activations, ~10us at 23.3 TB/s).  GEMM via
// v_wmma_f32_16x16x32_bf16 with the hi/lo bf16 split (Ahi*Whi + Ahi*Wlo +
// Alo*Whi) for ~fp32 accuracy.  W (1.18 MB) is repacked once into the native
// wave32 B-fragment layout so each lane's B operand is one aligned 32B load.
// Activations are streamed with non-temporal hints (dead after one use, and
// 226 MB > 192 MB L2) + global_prefetch for latency; W fragments stay
// regular-temporal so they live in WGP$/L2.
// 6 waves per block (2 waves per matrix, 32 rows) halves the W re-read
// traffic vs 16-row blocks while keeping 1536 waves in flight.
// ---------------------------------------------------------------------------

typedef __attribute__((ext_vector_type(16))) __bf16 v16bf;
typedef __attribute__((ext_vector_type(8)))  float  v8f;
typedef __attribute__((ext_vector_type(4)))  float  v4f;

#define EMB    128
#define KDIM   2304
#define KTILES (KDIM / 32)   // 72
#define NTILES (EMB / 16)    // 8
#define BATCH  8192
#define ALPHA_ 0.2f
#define FRAG   512           // bf16 elems per (ktile,ntile) fragment: 32 lanes * 16
#define ROWS_PER_BLOCK 32

__device__ __forceinline__ v8f wmma_bf16(v16bf a, v16bf b, v8f c) {
  // (neg_a, A, neg_b, B, c_mod, C, reuse_a, reuse_b)
  return __builtin_amdgcn_wmma_f32_16x16x32_bf16(false, a, false, b, (short)0, c,
                                                 false, false);
}

// --- Kernel 0: repack W (f32 [2304,128]) into bf16 hi/lo B-fragments --------
// Fragment layout (v_wmma 16x16x32 B operand, wave32):
//   lane l in [0,32): N = ntile*16 + (l&15); K = (l>>4)*8 + (e&7) + 16*(e>>3).
// Flat index i = ((kt*8 + nt)*32 + l)*16 + e.
__global__ void prep_w(const float* __restrict__ W,
                       unsigned short* __restrict__ whi_u,
                       unsigned short* __restrict__ wlo_u) {
  int i = blockIdx.x * blockDim.x + threadIdx.x;   // < KTILES*NTILES*FRAG = 294912
  __bf16* whi = (__bf16*)whi_u;
  __bf16* wlo = (__bf16*)wlo_u;
  int e  = i & 15;
  int l  = (i >> 4) & 31;
  int f  = i >> 9;
  int nt = f & 7;
  int kt = f >> 3;
  int k  = kt * 32 + ((l >> 4) * 8) + (e & 7) + ((e >> 3) * 16);
  int n  = nt * 16 + (l & 15);
  float x   = W[k * EMB + n];
  __bf16 h  = (__bf16)x;
  __bf16 lo = (__bf16)(x - (float)h);
  whi[i] = h;
  wlo[i] = lo;
}

// --- Kernel 1: fused 3-way GEMM + triplet distance --------------------------
// 192 threads = 6 waves; wave w: matrix = w>>1 (anchor/pos/neg), mtile = w&1.
__global__ void __launch_bounds__(192)
triplet_gemm(const float* __restrict__ Aanc, const float* __restrict__ Apos,
             const float* __restrict__ Aneg, const float* __restrict__ bias,
             const unsigned short* __restrict__ whi_u,
             const unsigned short* __restrict__ wlo_u,
             float* __restrict__ lossArr) {
  const __bf16* Whi = (const __bf16*)whi_u;
  const __bf16* Wlo = (const __bf16*)wlo_u;
  __shared__ float semb[3][ROWS_PER_BLOCK][EMB];   // 48 KB

  const int tid   = threadIdx.x;
  const int wave  = tid >> 5;
  const int lane  = tid & 31;
  const int mat   = wave >> 1;       // 0=anchor, 1=pos, 2=neg
  const int mtile = wave & 1;
  const float* X  = (mat == 0) ? Aanc : ((mat == 1) ? Apos : Aneg);

  const int rowBase = blockIdx.x * ROWS_PER_BLOCK + mtile * 16;
  const int m    = lane & 15;       // A row / C column-within-tile
  const int hsel = lane >> 4;
  const int off0 = hsel * 8;        // A-fragment K striping
  const float* rowp = X + (size_t)(rowBase + m) * KDIM + off0;

  v8f acc[NTILES];
  {
    v8f z = {0.f, 0.f, 0.f, 0.f, 0.f, 0.f, 0.f, 0.f};
#pragma unroll
    for (int nt = 0; nt < NTILES; ++nt) acc[nt] = z;
  }

  for (int kt = 0; kt < KTILES; ++kt) {
    // Prefetch the activation stream a few K-tiles ahead (global_prefetch_b8,
    // speculative: safe even past the row end).
    __builtin_prefetch(rowp + (kt + 6) * 32, 0, 3);

    // A fragment: per lane two contiguous 8-float runs (K off0..off0+7 and
    // K off0+16..off0+23 within the 32-wide K tile) -> four b128 NT loads.
    v4f q0 = __builtin_nontemporal_load((const v4f*)(rowp + kt * 32));
    v4f q1 = __builtin_nontemporal_load((const v4f*)(rowp + kt * 32 + 4));
    v4f q2 = __builtin_nontemporal_load((const v4f*)(rowp + kt * 32 + 16));
    v4f q3 = __builtin_nontemporal_load((const v4f*)(rowp + kt * 32 + 20));
    float xs[16] = {q0[0], q0[1], q0[2], q0[3], q1[0], q1[1], q1[2], q1[3],
                    q2[0], q2[1], q2[2], q2[3], q3[0], q3[1], q3[2], q3[3]};
    v16bf ah, al;
#pragma unroll
    for (int i = 0; i < 16; ++i) {
      float x  = xs[i];
      __bf16 h = (__bf16)x;
      ah[i] = h;
      al[i] = (__bf16)(x - (float)h);     // residual for the split GEMM
    }

    const __bf16* wb = Whi + ((size_t)kt * NTILES) * FRAG + lane * 16;
    const __bf16* wl = Wlo + ((size_t)kt * NTILES) * FRAG + lane * 16;
#pragma unroll
    for (int nt = 0; nt < NTILES; ++nt) {
      v16bf bh = *(const v16bf*)(wb + nt * FRAG);   // 32B aligned, temporal
      v16bf bl = *(const v16bf*)(wl + nt * FRAG);
      acc[nt] = wmma_bf16(ah, bh, acc[nt]);
      acc[nt] = wmma_bf16(ah, bl, acc[nt]);
      acc[nt] = wmma_bf16(al, bh, acc[nt]);
    }
  }

  // C/D layout: lane l -> N = l&15 ; VGPR r -> M = r + 8*(l>>4).
  const int mrow = mtile * 16 + 8 * hsel;
#pragma unroll
  for (int nt = 0; nt < NTILES; ++nt) {
    int n = nt * 16 + m;
    float bv = bias[n];
#pragma unroll
    for (int r = 0; r < 8; ++r)
      semb[mat][mrow + r][n] = acc[nt][r] + bv;
  }
  __syncthreads();

  if (tid < ROWS_PER_BLOCK) {
    float dap = 0.f, dan = 0.f;
#pragma unroll 8
    for (int n = 0; n < EMB; ++n) {
      float av = semb[0][tid][n];
      float dp = av - semb[1][tid][n];
      float dn = av - semb[2][tid][n];
      dap += dp * dp;
      dan += dn * dn;
    }
    float L = dap - dan + ALPHA_;
    __builtin_nontemporal_store(L > 0.f ? L : 0.f,
                                &lossArr[blockIdx.x * ROWS_PER_BLOCK + tid]);
  }
}

// --- Kernel 2: deterministic fixed-order mean reduction ---------------------
__global__ void reduce_loss(const float* __restrict__ lossArr, float* __restrict__ out) {
  __shared__ float s[256];
  float v = 0.f;
  for (int i = threadIdx.x; i < BATCH; i += 256) v += lossArr[i];
  s[threadIdx.x] = v;
  __syncthreads();
  for (int st = 128; st > 0; st >>= 1) {
    if (threadIdx.x < st) s[threadIdx.x] += s[threadIdx.x + st];
    __syncthreads();
  }
  if (threadIdx.x == 0) out[0] = s[0] * (1.0f / (float)BATCH);
}

extern "C" void kernel_launch(void* const* d_in, const int* in_sizes, int n_in,
                              void* d_out, int out_size, void* d_ws, size_t ws_size,
                              hipStream_t stream) {
  (void)in_sizes; (void)n_in; (void)out_size;
  const float* anc  = (const float*)d_in[0];
  const float* pos  = (const float*)d_in[1];
  const float* neg  = (const float*)d_in[2];
  const float* W    = (const float*)d_in[3];
  const float* bias = (const float*)d_in[4];

  // Workspace layout: Whi (589824 B) | Wlo (589824 B) | per-row loss (32 KB)
  char* ws = (char*)d_ws;
  unsigned short* whi = (unsigned short*)(ws);
  unsigned short* wlo = (unsigned short*)(ws + 589824);
  float* lossArr      = (float*)(ws + 2 * 589824);
  (void)ws_size;  // needs ~1.16 MB

  const int packElems = KTILES * NTILES * FRAG;        // 294912
  prep_w<<<packElems / 256, 256, 0, stream>>>(W, whi, wlo);
  triplet_gemm<<<BATCH / ROWS_PER_BLOCK, 192, 0, stream>>>(anc, pos, neg, bias,
                                                           whi, wlo, lossArr);
  reduce_loss<<<1, 256, 0, stream>>>(lossArr, (float*)d_out);
}
